// SpatialTransformerPooled3d_9792525435349
// MI455X (gfx1250) — matrix operand order repaired and checked by reading, not compiled, not run on hardware
//
#include <hip/hip_runtime.h>

typedef __attribute__((ext_vector_type(16))) _Float16 v16h;
typedef __attribute__((ext_vector_type(8)))  float    v8f;

#define CCH   64
#define TT    16
#define BATCH 128          // N*T
#define OUTD  2048

// ---- workspace layout (bytes) : ~42.1 MB total ----
#define Z1_ELEMS (BATCH*CCH*32*32)          // 8,388,608 floats
#define Z2_ELEMS (BATCH*CCH*16*16)          // 2,097,152 floats
#define Z1_OFF   ((size_t)0)
#define Z2_OFF   ((size_t)Z1_ELEMS*4)                    // 33,554,432
#define TOFF_OFF ((size_t)(Z2_OFF + (size_t)Z2_ELEMS*4)) // 41,943,040
#define TWT_OFF  ((size_t)(TOFF_OFF + (size_t)OUTD*3*8)) // + 49,152

// unaligned-safe 8-byte pair load (gfx1250 unaligned access mode -> b64)
struct __attribute__((packed)) F2 { float a, b; };

// ---------------- pooling pre-pass (HBM-bound, read x once) ----------------
__global__ void pool1_kernel(const float* __restrict__ x, float* __restrict__ z1) {
    int idx = blockIdx.x * 256 + threadIdx.x;
    if (idx >= Z1_ELEMS) return;
    int xw = idx & 31;
    int yw = (idx >> 5) & 31;
    int c  = (idx >> 10) & 63;
    int b  = idx >> 16;
    int n = b >> 4, t = b & 15;
    const float* src = x + ((size_t)((n * CCH + c) * TT + t) << 12); // *4096
    int y2 = yw * 2, x2 = xw * 2;
    float s = src[y2 * 64 + x2] + src[y2 * 64 + x2 + 1]
            + src[(y2 + 1) * 64 + x2] + src[(y2 + 1) * 64 + x2 + 1];
    z1[idx] = s * 0.25f;
}

__global__ void pool2_kernel(const float* __restrict__ z1, float* __restrict__ z2) {
    int idx = blockIdx.x * 256 + threadIdx.x;
    if (idx >= Z2_ELEMS) return;
    int xw = idx & 15;
    int yw = (idx >> 4) & 15;
    int bc = idx >> 8;
    const float* src = z1 + (size_t)bc * 1024;
    int y2 = yw * 2, x2 = xw * 2;
    float s = src[y2 * 32 + x2] + src[y2 * 32 + x2 + 1]
            + src[(y2 + 1) * 32 + x2] + src[(y2 + 1) * 32 + x2 + 1];
    z2[idx] = s * 0.25f;
}

// ---------------- tap precompute: grid is batch-invariant --------------
// Factorized form: sample = wy0*(wx0*z[r0] + wx1*z[r0+1])
//                        + wy1*(wx0*z[r1] + wx1*z[r1+1])
// with r0 = by*S+bx (bx,by clamped to [0,S-2]) so both 8-byte row-pairs are
// always in-bounds; edge validity is folded into the permuted wx/wy weights.
__global__ void taps_kernel(const float* __restrict__ grid,
                            int2* __restrict__ toff, float4* __restrict__ twt) {
    int tid = blockIdx.x * 64 + threadIdx.x;
    if (tid >= OUTD * 3) return;
    int o = tid / 3, m = tid % 3;
    float gx = fminf(fmaxf(grid[2 * o + 0], -1.f), 1.f);
    float gy = fminf(fmaxf(grid[2 * o + 1], -1.f), 1.f);
    int S = 64 >> m;
    float fx = ((gx + 1.f) * (float)S - 1.f) * 0.5f;   // align_corners=False
    float fy = ((gy + 1.f) * (float)S - 1.f) * 0.5f;
    float fx0 = floorf(fx), fy0 = floorf(fy);
    float wx = fx - fx0, wy = fy - fy0;
    int x0 = (int)fx0, y0 = (int)fy0;

    // x direction: taps at x0 (1-wx) and x0+1 (wx); pair base bx
    int bx = min(max(x0, 0), S - 2);
    float wx0 = 0.f, wx1 = 0.f;
    if (x0 >= 0 && x0 < S)          { if (x0 == bx)     wx0 += 1.f - wx; else wx1 += 1.f - wx; }
    if (x0 + 1 >= 0 && x0 + 1 < S)  { if (x0 + 1 == bx) wx0 += wx;       else wx1 += wx; }
    // y direction
    int by = min(max(y0, 0), S - 2);
    float wy0 = 0.f, wy1 = 0.f;
    if (y0 >= 0 && y0 < S)          { if (y0 == by)     wy0 += 1.f - wy; else wy1 += 1.f - wy; }
    if (y0 + 1 >= 0 && y0 + 1 < S)  { if (y0 + 1 == by) wy0 += wy;       else wy1 += wy; }

    int r0 = by * S + bx;
    toff[tid] = make_int2(r0, r0 + S);
    twt[tid]  = make_float4(wx0, wx1, wy0, wy1);
}

// ---------------- main gather + WMMA contraction ----------------
// 512 blocks x 256 threads = 4096 waves. A quad of waves shares one 16x16
// (batch x output) tile: each wave handles 4 of the 16 o-columns
// (4 o * 6 K-chunks = 24 WMMAs), partial D reduced through LDS.
__global__ __launch_bounds__(256) void stp3d_wmma_kernel(
    const float*  __restrict__ x,    const float* __restrict__ feat,
    const float*  __restrict__ bias, const float* __restrict__ z1,
    const float*  __restrict__ z2,   const int2*  __restrict__ toff,
    const float4* __restrict__ twt,  float*       __restrict__ out)
{
    __shared__ float red[2][3][8][32];
    int lane = threadIdx.x & 31;
    int wave = threadIdx.x >> 5;
    int tile = wave >> 2;          // 0..1 : which 16-wide o-tile in this block
    int quad = wave & 3;           // 4-way split of the 16 o-columns
    int b_tile = blockIdx.x >> 6;  // 0..7  (== n, since T == 16)
    int o_pair = blockIdx.x & 63;  // 0..63
    int o_tile = o_pair * 32 + tile * 16;

    int ln16 = lane & 15;
    int g    = lane >> 4;

    // B-matrix (32x16 f16) layout per ISA: lane row N=ln16; half h -> K = (h<8 ? h : h+8) + 8*(lane/16).
    v16h bfull[6];
    int colo = o_tile + ln16;
#pragma unroll
    for (int j = 0; j < 6; ++j) {
#pragma unroll
        for (int h = 0; h < 16; ++h) {
            int k = 32 * j + ((h < 8) ? h : h + 8) + 8 * g;
            bfull[j][h] = (_Float16)feat[k * OUTD + colo];
        }
    }

    // Per-lane source bases. A row M = ln16 -> b = b_tile*16 + ln16,
    // i.e. n = b_tile, t = ln16 (x layout is (N,C,T,W,H)).
    size_t base0 = (size_t)b_tile * 4194304 + (size_t)ln16 * 4096; // x, c-stride 65536
    int brow = b_tile * 16 + ln16;
    size_t base1 = (size_t)brow * 65536;   // z1: (b*64+c)*1024
    size_t base2 = (size_t)brow * 16384;   // z2: (b*64+c)*256

    v8f acc = {0.f, 0.f, 0.f, 0.f, 0.f, 0.f, 0.f, 0.f};
    v16h bzero = {};

#pragma unroll 1
    for (int ol = quad * 4; ol < quad * 4 + 4; ++ol) {
        int o = o_tile + ol;
#pragma unroll
        for (int m = 0; m < 3; ++m) {
            int2   to = toff[o * 3 + m];
            float4 tw = twt [o * 3 + m];
            const float* zp  = (m == 0) ? x : (m == 1) ? z1 : z2;
            size_t base      = (m == 0) ? base0 : (m == 1) ? base1 : base2;
            int    cstride   = (m == 0) ? 65536 : (m == 1) ? 1024 : 256;
#pragma unroll
            for (int j2 = 0; j2 < 2; ++j2) {
                int j = m * 2 + j2;
                v16h a;
                // A-matrix (16x32 f16) layout per ISA: lane row M=ln16;
                // half h -> K = (v<4 ? 2v+w : 16+2(v-4)+w) + 8*(lane/16).
#pragma unroll
                for (int h = 0; h < 16; ++h) {
                    int v = h >> 1, wsel = h & 1;
                    int k = ((v < 4) ? (2 * v + wsel) : (16 + 2 * (v - 4) + wsel)) + 8 * g;
                    int c = 32 * j2 + k;               // channel within scale m
                    const float* p = zp + base + (size_t)c * (size_t)cstride;
                    F2 u0 = *(const F2*)(p + to.x);    // row y0 pair (b64 gather)
                    F2 u1 = *(const F2*)(p + to.y);    // row y1 pair
                    float s = tw.z * (tw.x * u0.a + tw.y * u0.b)
                            + tw.w * (tw.x * u1.a + tw.y * u1.b);
                    a[h] = (_Float16)s;
                }
                // mask feat chunk to this o's column only
                v16h bm = (ln16 == ol) ? bfull[j] : bzero;
                acc = __builtin_amdgcn_wmma_f32_16x16x32_f16(
                        false, a, false, bm, (short)0, acc, false, false);
            }
        }
    }

    // reduce the quad's partial D tiles, add bias, store.
    if (quad != 0) {
#pragma unroll
        for (int r = 0; r < 8; ++r) red[tile][quad - 1][r][lane] = acc[r];
    }
    __syncthreads();
    if (quad == 0) {
        float bv = bias[o_tile + ln16];
#pragma unroll
        for (int r = 0; r < 8; ++r) {
            float vsum = acc[r] + red[tile][0][r][lane]
                       + red[tile][1][r][lane] + red[tile][2][r][lane];
            int trow = r + 8 * g;                       // D row M = r + 8*(lane/16)
            out[(size_t)(b_tile * 16 + trow) * OUTD + o_tile + ln16] = vsum + bv;
        }
    }
}

extern "C" void kernel_launch(void* const* d_in, const int* in_sizes, int n_in,
                              void* d_out, int out_size, void* d_ws, size_t ws_size,
                              hipStream_t stream) {
    const float* x    = (const float*)d_in[0];
    const float* grid = (const float*)d_in[1];
    const float* feat = (const float*)d_in[2];
    const float* bias = (const float*)d_in[3];
    char* ws = (char*)d_ws;
    float*  z1   = (float*)(ws + Z1_OFF);
    float*  z2   = (float*)(ws + Z2_OFF);
    int2*   toff = (int2*) (ws + TOFF_OFF);
    float4* twt  = (float4*)(ws + TWT_OFF);
    float* out = (float*)d_out;

    pool1_kernel<<<(Z1_ELEMS + 255) / 256, 256, 0, stream>>>(x, z1);
    pool2_kernel<<<(Z2_ELEMS + 255) / 256, 256, 0, stream>>>(z1, z2);
    taps_kernel <<<(OUTD * 3 + 63) / 64,   64,  0, stream>>>(grid, toff, twt);
    stp3d_wmma_kernel<<<512, 256, 0, stream>>>(x, feat, bias, z1, z2, toff, twt, out);
}
